// MADDPG_85272280695443
// MI455X (gfx1250) — matrix operand order repaired and checked by reading, not compiled, not run on hardware
//
#include <hip/hip_runtime.h>
#include <hip/hip_bf16.h>

typedef __attribute__((ext_vector_type(16))) _Float16 v16h;
typedef __attribute__((ext_vector_type(8)))  _Float16 v8h;
typedef __attribute__((ext_vector_type(8)))  float    v8f;

#define RADIUS2 (0.15f * 0.15f)
#define L1C 0.1f
#define L2C 2.0f

// Wave-local LDS ordering: all LDS staging in this kernel is wave-private and
// all cross-lane exchange is within one wave32.  CDNA5 LDS ops from the same
// wave complete in order (DScnt), so a wave-local s_wait_dscnt + compiler
// memory barrier replaces __syncthreads() and lets the 8 waves run free.
#define WAVE_LDS_FENCE() asm volatile("s_wait_dscnt 0x0" ::: "memory")

// ---------------------------------------------------------------------------
// Weight repack: build V_WMMA_F32_16X16X32_F16 B-fragments in global scratch.
// Lane 0-15 holds column N = lane, K = kt*32 + 0..7,16..23;
// lanes 16-31 hold N = lane-16, K = kt*32 + 8..15,24..31.  32 lanes x 16 halves.
//   W1f: 2 kt x 4 nt (K=39 padded to 64, N=64) -> 4096 halves @ wp+0
//   W2f: 2 kt x 4 nt (K=64, N=64)              -> 4096 halves @ wp+4096
//   W3f: 2 kt x 1 nt (K=64, N=5 padded to 16)  -> 1024 halves @ wp+8192
// ---------------------------------------------------------------------------
__global__ void maddpg_pack_weights(const float* __restrict__ W1,
                                    const float* __restrict__ W2,
                                    const float* __restrict__ W3,
                                    _Float16* __restrict__ wp) {
    int t = blockIdx.x * blockDim.x + threadIdx.x;
    for (int e = t; e < 9216; e += blockDim.x * gridDim.x) {
        const float* W; int K, Nc, ntiles, base;
        if (e < 4096)      { W = W1; K = 39; Nc = 64; ntiles = 4; base = 0;    }
        else if (e < 8192) { W = W2; K = 64; Nc = 64; ntiles = 4; base = 4096; }
        else               { W = W3; K = 64; Nc = 5;  ntiles = 1; base = 8192; }
        int le   = e - base;
        int h    = le & 15;
        int lane = (le >> 4) & 31;
        int frag = le >> 9;               // kt*ntiles + nt
        int kt   = frag / ntiles;
        int nt   = frag % ntiles;
        int n    = nt * 16 + (lane & 15);
        int k    = kt * 32 + ((lane < 16) ? 0 : 8) + (h < 8 ? h : h + 8);
        float v  = (k < K && n < Nc) ? W[k * Nc + n] : 0.0f;
        wp[e] = (_Float16)v;
    }
}

// ---------------------------------------------------------------------------
// Fused observation-build + 3-layer MLP + action head.
// Block = 256 threads = 8 independent waves; wave owns 32 samples (2 M-tiles).
// ---------------------------------------------------------------------------
__global__ __launch_bounds__(256) void maddpg_fused(
    const float* __restrict__ x,    // [B,20,7]
    const float* __restrict__ b1,   // [64]
    const float* __restrict__ b2,   // [64]
    const float* __restrict__ b3,   // [5]
    const int*   __restrict__ idx,  // [8]
    const _Float16* __restrict__ wp,
    float* __restrict__ out)        // [B,2]
{
    __shared__ __align__(16) _Float16 lds_obs[8][32][72]; // obs, K padded to 64
    __shared__ __align__(16) _Float16 lds_h  [8][16][72]; // hidden re-layout
    __shared__ __align__(16) float    lds_r  [8][16][8];  // final 5 outputs

    const int tid   = threadIdx.x;
    const int wave  = tid >> 5;
    const int lane  = tid & 31;
    const int lhalf = lane & 15;
    const bool hi   = lane >= 16;

    const int s = blockIdx.x * 256 + wave * 32 + lane;   // this lane's sample
    const float* xs = x + (size_t)s * 140;

    // ---- build 39-dim observation in registers ----
    float obs[40];
    {
        float4 p0 = *(const float4*)(xs);                // 560B stride -> 16B aligned
        obs[0] = p0.x; obs[1] = p0.y; obs[2] = p0.z; obs[3] = p0.w; obs[4] = xs[4];
        const float px = p0.x, py = p0.y;
        #pragma unroll
        for (int i = 0; i < 3; i++) {
            const float* xr = xs + idx[i] * 7;
            obs[5 + 3*i] = xr[5] + xr[0] - px;
            obs[6 + 3*i] = xr[6] + xr[1] - py;
            obs[7 + 3*i] = 0.0f;
        }
        #pragma unroll
        for (int j = 0; j < 5; j++) {
            const float* xr = xs + (idx[3 + j] + 10) * 7;
            float dx = xr[0] - px, dy = xr[1] - py;
            float mm = ((dx*dx + dy*dy) < RADIUS2) ? 1.0f : 0.0f;
            obs[14 + 5*j + 0] = dx * mm;
            obs[14 + 5*j + 1] = dy * mm;
            obs[14 + 5*j + 2] = xr[2] * mm;
            obs[14 + 5*j + 3] = xr[3] * mm;
            obs[14 + 5*j + 4] = xr[4] * mm;
        }
        obs[39] = 0.0f;
    }
    // ---- stash obs row as f16 in LDS (K zero-padded to 64) ----
    {
        _Float16* myrow = &lds_obs[wave][lane][0];
        #pragma unroll
        for (int c = 0; c < 8; c++) {
            v8h t;
            #pragma unroll
            for (int i2 = 0; i2 < 8; i2++) {
                int k = c * 8 + i2;
                t[i2] = (k < 39) ? (_Float16)obs[k] : (_Float16)0.0f;
            }
            *(v8h*)(myrow + c * 8) = t;
        }
    }
    WAVE_LDS_FENCE();

    const _Float16* w1f = wp;
    const _Float16* w2f = wp + 4096;
    const _Float16* w3f = wp + 8192;

    for (int mt = 0; mt < 2; mt++) {
        // ================= Layer 1: obs(39->64) @ W1, ReLU =================
        v8f acc[4] = {};
        #pragma unroll
        for (int kt = 0; kt < 2; kt++) {
            // A fragment (16x32 f16): lane<16 -> K 0..7 & 16..23 of row lhalf
            const _Float16* ar = &lds_obs[wave][mt * 16 + lhalf][0];
            int off = kt * 32 + (hi ? 8 : 0);
            v8h alo = *(const v8h*)(ar + off);
            v8h ahi = *(const v8h*)(ar + off + 16);
            v16h a = __builtin_shufflevector(alo, ahi,
                0,1,2,3,4,5,6,7,8,9,10,11,12,13,14,15);
            #pragma unroll
            for (int nt = 0; nt < 4; nt++) {
                v16h b = *(const v16h*)(w1f + (size_t)((kt * 4 + nt) * 32 + lane) * 16);
                acc[nt] = __builtin_amdgcn_wmma_f32_16x16x32_f16(
                    false, a, false, b, (short)0, acc[nt], false, false);
            }
        }
        #pragma unroll
        for (int nt = 0; nt < 4; nt++) {
            float bn = b1[nt * 16 + lhalf];
            #pragma unroll
            for (int r = 0; r < 8; r++) {
                int row = hi ? (8 + r) : r;
                float v = acc[nt][r] + bn;
                lds_h[wave][row][nt * 16 + lhalf] = (_Float16)(v > 0.0f ? v : 0.0f);
            }
        }
        WAVE_LDS_FENCE();

        // ================= Layer 2: h(64->64) @ W2, ReLU ===================
        v8f acc2[4] = {};
        #pragma unroll
        for (int kt = 0; kt < 2; kt++) {
            const _Float16* ar = &lds_h[wave][lhalf][0];
            int off = kt * 32 + (hi ? 8 : 0);
            v8h alo = *(const v8h*)(ar + off);
            v8h ahi = *(const v8h*)(ar + off + 16);
            v16h a = __builtin_shufflevector(alo, ahi,
                0,1,2,3,4,5,6,7,8,9,10,11,12,13,14,15);
            #pragma unroll
            for (int nt = 0; nt < 4; nt++) {
                v16h b = *(const v16h*)(w2f + (size_t)((kt * 4 + nt) * 32 + lane) * 16);
                acc2[nt] = __builtin_amdgcn_wmma_f32_16x16x32_f16(
                    false, a, false, b, (short)0, acc2[nt], false, false);
            }
        }
        WAVE_LDS_FENCE();   // layer-2 A reads complete before overwriting lds_h
        #pragma unroll
        for (int nt = 0; nt < 4; nt++) {
            float bn = b2[nt * 16 + lhalf];
            #pragma unroll
            for (int r = 0; r < 8; r++) {
                int row = hi ? (8 + r) : r;
                float v = acc2[nt][r] + bn;
                lds_h[wave][row][nt * 16 + lhalf] = (_Float16)(v > 0.0f ? v : 0.0f);
            }
        }
        WAVE_LDS_FENCE();

        // ================= Layer 3: h(64->5) @ W3 ==========================
        v8f acc3 = {};
        #pragma unroll
        for (int kt = 0; kt < 2; kt++) {
            const _Float16* ar = &lds_h[wave][lhalf][0];
            int off = kt * 32 + (hi ? 8 : 0);
            v8h alo = *(const v8h*)(ar + off);
            v8h ahi = *(const v8h*)(ar + off + 16);
            v16h a = __builtin_shufflevector(alo, ahi,
                0,1,2,3,4,5,6,7,8,9,10,11,12,13,14,15);
            v16h b = *(const v16h*)(w3f + (size_t)(kt * 32 + lane) * 16);
            acc3 = __builtin_amdgcn_wmma_f32_16x16x32_f16(
                false, a, false, b, (short)0, acc3, false, false);
        }
        // scatter r1..r4 (+bias) to LDS for the per-row head
        if (lhalf >= 1 && lhalf <= 4) {
            float bn = b3[lhalf];
            #pragma unroll
            for (int r = 0; r < 8; r++) {
                int row = hi ? (8 + r) : r;
                lds_r[wave][row][lhalf] = acc3[r] + bn;
            }
        }
        WAVE_LDS_FENCE();

        // ================= head: res = clamp((r1-r2, r3-r4) * 0.1) =========
        if (lane < 16) {
            float r1 = lds_r[wave][lane][1];
            float r2 = lds_r[wave][lane][2];
            float r3 = lds_r[wave][lane][3];
            float r4 = lds_r[wave][lane][4];
            float res0 = (r1 - r2) * L1C;
            float res1 = (r3 - r4) * L1C;
            if (fabsf(res0) > 1.0f) res0 = (res0 > 0.0f) ? L2C : -L2C;
            if (fabsf(res1) > 1.0f) res1 = (res1 > 0.0f) ? L2C : -L2C;
            int so = blockIdx.x * 256 + wave * 32 + mt * 16 + lane;
            *(float2*)(out + (size_t)so * 2) = make_float2(res0, res1);
        }
        WAVE_LDS_FENCE();   // lds_h / lds_r reused by next M-tile
    }
}

extern "C" void kernel_launch(void* const* d_in, const int* in_sizes, int n_in,
                              void* d_out, int out_size, void* d_ws, size_t ws_size,
                              hipStream_t stream) {
    const float* x  = (const float*)d_in[0];
    const float* W1 = (const float*)d_in[1];
    const float* b1 = (const float*)d_in[2];
    const float* W2 = (const float*)d_in[3];
    const float* b2 = (const float*)d_in[4];
    const float* W3 = (const float*)d_in[5];
    const float* b3 = (const float*)d_in[6];
    const int*   idx = (const int*)d_in[7];
    float* out = (float*)d_out;

    _Float16* wp = (_Float16*)d_ws;   // 18432 B of packed f16 weight fragments

    maddpg_pack_weights<<<36, 256, 0, stream>>>(W1, W2, W3, wp);

    int nsamples = in_sizes[0] / 140;          // B = 1048576
    int nblocks  = nsamples / 256;             // 256 samples per block
    maddpg_fused<<<nblocks, 256, 0, stream>>>(x, b1, b2, b3, idx, wp, out);
}
